// AttentionLayer_55551107006649
// MI455X (gfx1250) — matrix operand order repaired and checked by reading, not compiled, not run on hardware
//
#include <hip/hip_runtime.h>
#include <math.h>

// ---- WMMA vector types (gfx1250, wave32) ----
typedef __attribute__((ext_vector_type(16))) __bf16 v16bf;
typedef __attribute__((ext_vector_type(8)))  float  v8f;

#define B_   32
#define U_   256
#define DIN  512
#define DOUT 64
#define C_   1024
#define DV   64
#define CC   128   // C-chunk
#define KC   128   // K-chunk for the Q GEMM

#if defined(__has_builtin)
#if __has_builtin(__builtin_amdgcn_cvt_pk_bf16_f32)
#define HAS_PK_BF16 1
#endif
#endif

__device__ __forceinline__ unsigned int pack2bf(float lo, float hi) {
#ifdef HAS_PK_BF16
    typedef __attribute__((ext_vector_type(2))) __bf16 v2bf;
    union { v2bf v; unsigned int w; } un;
    un.v = __builtin_amdgcn_cvt_pk_bf16_f32(lo, hi);   // v_cvt_pk_bf16_f32
    return un.w;
#else
    union { __bf16 h[2]; unsigned int w; } un;
    un.h[0] = (__bf16)lo;
    un.h[1] = (__bf16)hi;
    return un.w;
#endif
}

// Fragment tiles live in LDS already in ISA lane order: tile = [32 lanes][8 dwords].
// Loading a fragment is two 16-byte LDS reads per lane.
__device__ __forceinline__ v16bf load_frag(const unsigned int* tile, int lane) {
    union { v16bf v; uint4 q[2]; } f;
    const uint4* p = (const uint4*)(tile + (lane << 3));
    f.q[0] = p[0];
    f.q[1] = p[1];
    return f.v;
}

// A-tile (16x32, row-major source, k contiguous): element (m,k):
//   lane = (m&15) + 16*((k>>4)&1) ; dword i = (k&15)>>1
// B-tile (32x16): element (k,n): lane = n + 16*(k>>4) ; dword i = (k&15)>>1

#define WMMA_BF16(A, B, Cacc) \
    __builtin_amdgcn_wmma_f32_16x16x32_bf16(false, (A), false, (B), (short)0, (Cacc), false, false)

// =====================================================================
// Fused kernel: per unit u (grid = U, block = 128 = 4 waves)
//   phase 0 : Q = attention[32x512] x W[u][512x64]        (WMMA, LDS only)
//   pass 1  : logits = Q·keysᵀ/τ, mask, row max           (WMMA)
//   pass 2  : e=exp, Z, Zr, O = (e·r)·values              (WMMA)
//   pass 3  : weights = e/Z, outputs = O/Zr, rewards=Zr/Z
// d_out layout: outputs[B,U,DV] | weights[B,U,C] | rewards[B,U]
// =====================================================================
__global__ __launch_bounds__(128)
void attention_unit_fused(const float* __restrict__ attn,         // [B][DIN]
                          const float* __restrict__ W,            // [U][DIN][DOUT]
                          const float* __restrict__ temperature,  // [U]
                          const float* __restrict__ keys,         // [U][C][DOUT]
                          const float* __restrict__ values,       // [U][C][DV]
                          const float* __restrict__ rmem,         // [U][C]
                          const int*   __restrict__ mask,         // [B][U][C]
                          float* __restrict__ out)
{
    const int u    = blockIdx.x;
    const int tid  = threadIdx.x;
    const int lane = tid & 31;
    const int wave = tid >> 5;
    const int half = lane >> 4;
    const int l16  = lane & 15;
    const int rE   = tid & 31;   // elementwise: row (batch)
    const int qE   = tid >> 5;   // elementwise: 32-col quarter == one K-tile

    const size_t W_OFF = (size_t)B_ * U_ * DV;
    const size_t R_OFF = W_OFF + (size_t)B_ * U_ * C_;

    // Phase-0 operand buffers alias the attention-phase buffers (both 24 KB).
    __shared__ __align__(16) union SMem {
        struct { unsigned int AsF[8][32][8]; unsigned int WsF[16][32][8]; } g;  // Q GEMM
        struct { unsigned int KVF[16][32][8]; unsigned int EsF[8][32][8]; } a;  // attention
    } sm;
    __shared__ __align__(16) unsigned int QsF[4][32][8];   // [mt*2+kt]  4 KB
    __shared__ __align__(16) float Ls[B_][CC + 4];         // 16.5 KB staging
    __shared__ __align__(16) float rs[CC];
    __shared__ float redA[4][B_], redB[4][B_];
    __shared__ float rowMax[B_], rowInvZ[B_], rowZr[B_];

    const float invT = 1.0f / temperature[u];

    // ================= Phase 0: Q = A x W[u] =================
    {
        v8f accQ0 = {};  // M rows 0..15, N cols wave*16..+15
        v8f accQ1 = {};  // M rows 16..31
        for (int k0 = 0; k0 < DIN; k0 += KC) {
            // attention chunk: 32 x 128, pairs along k (float2 loads)
            for (int p = tid; p < B_ * (KC / 2); p += 128) {
                int m = p >> 6, k = (p & 63) << 1;
                float2 x = *(const float2*)&attn[m * DIN + k0 + k];
                int kk = k & 31;
                int ln = (m & 15) + ((kk >> 4) << 4);
                int i  = (kk & 15) >> 1;
                sm.g.AsF[((m >> 4) << 2) + (k >> 5)][ln][i] = pack2bf(x.x, x.y);
            }
            // W chunk: 128 x 64; 2 k-rows x 2 o-cols per thread
            for (int p = tid; p < (KC / 2) * (DOUT / 2); p += 128) {
                int o = (p & 31) << 1, k = (p >> 5) << 1;
                size_t g = ((size_t)u * DIN + k0 + k) * DOUT + o;
                float2 r0 = *(const float2*)&W[g];
                float2 r1 = *(const float2*)&W[g + DOUT];
                int kk = k & 31;
                int tile = ((o >> 4) << 2) + (k >> 5);
                int ln   = ((kk >> 4) << 4) + (o & 15);
                int i    = (kk & 15) >> 1;
                sm.g.WsF[tile][ln    ][i] = pack2bf(r0.x, r1.x);
                sm.g.WsF[tile][ln + 1][i] = pack2bf(r0.y, r1.y);
            }
            if (k0 + KC < DIN)   // prefetch next W chunk (global_prefetch_b8)
                __builtin_prefetch(W + ((size_t)u * DIN + k0 + KC + tid) * DOUT, 0, 1);
            __syncthreads();
            #pragma unroll
            for (int kt = 0; kt < 4; ++kt) {
                v16bf bf = load_frag(&sm.g.WsF[(wave << 2) + kt][0][0], lane);
                v16bf a0 = load_frag(&sm.g.AsF[kt][0][0], lane);
                v16bf a1 = load_frag(&sm.g.AsF[4 + kt][0][0], lane);
                accQ0 = WMMA_BF16(a0, bf, accQ0);
                accQ1 = WMMA_BF16(a1, bf, accQ1);
            }
            __syncthreads();
        }
        // spill D-layout accumulators to LDS staging (f32), then repack to A-fragments
        #pragma unroll
        for (int i = 0; i < 8; ++i) {
            int m0 = i + 8 * half;
            Ls[m0     ][wave * 16 + l16] = accQ0[i];
            Ls[16 + m0][wave * 16 + l16] = accQ1[i];
        }
        __syncthreads();
        for (int p = tid; p < B_ * (DOUT / 2); p += 128) {
            int m = p >> 5, k = (p & 31) << 1;
            float2 x = *(const float2*)&Ls[m][k];
            int kk = k & 31;
            int ln = (m & 15) + ((kk >> 4) << 4);
            int i  = (kk & 15) >> 1;
            QsF[((m >> 4) << 1) + (k >> 5)][ln][i] = pack2bf(x.x, x.y);
        }
        __syncthreads();
    }

    // Q fragments are invariant over all C-chunks: keep them in VGPRs.
    v16bf qf[2][2];
    #pragma unroll
    for (int mt = 0; mt < 2; ++mt)
        #pragma unroll
        for (int kt = 0; kt < 2; ++kt)
            qf[mt][kt] = load_frag(&QsF[(mt << 1) + kt][0][0], lane);

    // ================= Pass 1: logits + mask + row max =================
    float rmax = -INFINITY;
    for (int c0 = 0; c0 < C_; c0 += CC) {
        // keys chunk: row-major, k contiguous -> B fragments [nt*2+kt]
        for (int p = tid; p < CC * (DOUT / 2); p += 128) {
            int row = p >> 5, k = (p & 31) << 1;
            float2 x = *(const float2*)&keys[((size_t)u * C_ + c0 + row) * DOUT + k];
            int kk = k & 31;
            int ln = (row & 15) + ((kk >> 4) << 4);
            int i  = (kk & 15) >> 1;
            sm.a.KVF[((row >> 4) << 1) + (k >> 5)][ln][i] = pack2bf(x.x, x.y);
        }
        if (c0 + CC < C_)
            __builtin_prefetch(keys + ((size_t)u * C_ + c0 + CC + tid) * DOUT, 0, 1);
        __syncthreads();
        #pragma unroll
        for (int tn = 0; tn < 2; ++tn) {
            const int nt = wave * 2 + tn;
            v16bf b0 = load_frag(&sm.a.KVF[(nt << 1)    ][0][0], lane);
            v16bf b1 = load_frag(&sm.a.KVF[(nt << 1) + 1][0][0], lane);
            #pragma unroll
            for (int mt = 0; mt < 2; ++mt) {
                v8f acc = {};
                acc = WMMA_BF16(qf[mt][0], b0, acc);
                acc = WMMA_BF16(qf[mt][1], b1, acc);
                #pragma unroll
                for (int i = 0; i < 8; ++i)
                    Ls[mt * 16 + i + 8 * half][nt * 16 + l16] = acc[i];
            }
        }
        __syncthreads();
        {   // scale + mask + running row max; vec4 everywhere
            const size_t wbase = W_OFF + ((size_t)rE * U_ + u) * C_ + c0 + qE * 32;
            const size_t mbase = ((size_t)rE * U_ + u) * C_ + c0 + qE * 32;
            const float4* lsv = (const float4*)&Ls[rE][qE * 32];
            const int4*   mv  = (const int4*)(mask + mbase);
            float4*       ov  = (float4*)(out + wbase);
            #pragma unroll
            for (int c4 = 0; c4 < 8; ++c4) {
                float4 l = lsv[c4];
                int4   mk = mv[c4];
                float4 r;
                r.x = mk.x ? l.x * invT : -INFINITY;
                r.y = mk.y ? l.y * invT : -INFINITY;
                r.z = mk.z ? l.z * invT : -INFINITY;
                r.w = mk.w ? l.w * invT : -INFINITY;
                ov[c4] = r;                       // L2-resident scratch
                rmax = fmaxf(rmax, fmaxf(fmaxf(r.x, r.y), fmaxf(r.z, r.w)));
            }
        }
        __syncthreads();
    }
    redA[qE][rE] = rmax;
    __syncthreads();
    if (tid < B_)
        rowMax[tid] = fmaxf(fmaxf(redA[0][tid], redA[1][tid]),
                            fmaxf(redA[2][tid], redA[3][tid]));
    __syncthreads();

    // ================= Pass 2: exp, Z, Zr, O = (e*r) x values =================
    v8f o[2] = {};
    float zq = 0.f, zrq = 0.f;
    for (int c0 = 0; c0 < C_; c0 += CC) {
        // values chunk: 2 c-rows x 2 v-cols per thread -> B fragments [kt*4+nt]
        for (int p = tid; p < (CC / 2) * (DOUT / 2); p += 128) {
            int v = (p & 31) << 1, c = (p >> 5) << 1;
            size_t g = ((size_t)u * C_ + c0 + c) * DOUT + v;
            float2 r0 = *(const float2*)&values[g];
            float2 r1 = *(const float2*)&values[g + DOUT];
            int kk = c & 31;
            int tile = ((c >> 5) << 2) + (v >> 4);
            int ln   = ((kk >> 4) << 4) + (v & 15);
            int i    = (kk & 15) >> 1;
            sm.a.KVF[tile][ln    ][i] = pack2bf(r0.x, r1.x);
            sm.a.KVF[tile][ln + 1][i] = pack2bf(r0.y, r1.y);
        }
        rs[tid] = rmem[(size_t)u * C_ + c0 + tid];
        if (c0 + CC < C_)
            __builtin_prefetch(values + ((size_t)u * C_ + c0 + CC + tid) * DOUT, 0, 1);
        __syncthreads();
        {   // exp + accumulate Z/Zr + pack A operand; quarter qE == K-tile qE
            const float m = rowMax[rE];
            const size_t wbase = W_OFF + ((size_t)rE * U_ + u) * C_ + c0 + qE * 32;
            float4*       ov = (float4*)(out + wbase);
            const float4* rv = (const float4*)&rs[qE * 32];
            const int tile = ((rE >> 4) << 2) + qE;
            const int lnlo = rE & 15;
            #pragma unroll
            for (int c4 = 0; c4 < 8; ++c4) {
                float4 l = ov[c4];
                float4 e;
                e.x = __expf(l.x - m);            // exp(-inf - m) = 0 for masked
                e.y = __expf(l.y - m);
                e.z = __expf(l.z - m);
                e.w = __expf(l.w - m);
                ov[c4] = e;
                zq += (e.x + e.y) + (e.z + e.w);
                float4 r = rv[c4];
                float er0 = e.x * r.x, er1 = e.y * r.y;
                float er2 = e.z * r.z, er3 = e.w * r.w;
                zrq += (er0 + er1) + (er2 + er3);
                int kk = c4 << 2;                 // k within this 32-tile (mult of 4)
                int ln = lnlo + ((kk >> 4) << 4);
                int i  = (kk & 15) >> 1;   // even; i,i+1 pair
                uint2 pk;
                pk.x = pack2bf(er0, er1);
                pk.y = pack2bf(er2, er3);
                *(uint2*)&sm.a.EsF[tile][ln][i] = pk;  // ds_store_b64
            }
        }
        __syncthreads();
        #pragma unroll
        for (int kt = 0; kt < 4; ++kt) {
            v16bf b  = load_frag(&sm.a.KVF[(kt << 2) + wave][0][0], lane);
            v16bf a0 = load_frag(&sm.a.EsF[kt    ][0][0], lane);
            v16bf a1 = load_frag(&sm.a.EsF[4 + kt][0][0], lane);
            o[0] = WMMA_BF16(a0, b, o[0]);
            o[1] = WMMA_BF16(a1, b, o[1]);
        }
        __syncthreads();
    }
    redA[qE][rE] = zq;
    redB[qE][rE] = zrq;
    __syncthreads();

    // ================= Finalize =================
    if (tid < B_) {
        float z  = redA[0][tid] + redA[1][tid] + redA[2][tid] + redA[3][tid];
        float zr = redB[0][tid] + redB[1][tid] + redB[2][tid] + redB[3][tid];
        rowInvZ[tid] = 1.0f / z;
        rowZr[tid]   = zr;
        out[R_OFF + (size_t)tid * U_ + u] = zr / z;     // rewards
    }
    __threadfence();
    __syncthreads();

    #pragma unroll
    for (int mt = 0; mt < 2; ++mt) {
        #pragma unroll
        for (int i = 0; i < 8; ++i) {
            int m = mt * 16 + i + 8 * half;
            out[((size_t)m * U_ + u) * DV + wave * 16 + l16] = o[mt][i] / rowZr[m];
        }
    }

    // ============ Pass 3: weights = e / Z (vec4, 256 contiguous floats/thread) ============
    {
        const float iz = rowInvZ[rE];
        float4* p = (float4*)(out + W_OFF + ((size_t)rE * U_ + u) * C_ + qE * 256);
        #pragma unroll 4
        for (int c = 0; c < 64; ++c) {
            float4 x = p[c];
            x.x *= iz; x.y *= iz; x.z *= iz; x.w *= iz;
            p[c] = x;
        }
    }
}

extern "C" void kernel_launch(void* const* d_in, const int* in_sizes, int n_in,
                              void* d_out, int out_size, void* d_ws, size_t ws_size,
                              hipStream_t stream) {
    (void)in_sizes; (void)n_in; (void)out_size; (void)d_ws; (void)ws_size;
    const float* attn   = (const float*)d_in[0];
    const float* W      = (const float*)d_in[1];
    const float* temp   = (const float*)d_in[2];
    const float* keys   = (const float*)d_in[3];
    const float* values = (const float*)d_in[4];
    const float* rmem   = (const float*)d_in[5];
    const int*   mask   = (const int*)d_in[6];
    float* out = (float*)d_out;

    attention_unit_fused<<<U_, 128, 0, stream>>>(attn, W, temp, keys, values,
                                                 rmem, mask, out);
}
